// Qwen2MoeSparseMoeBlock_62483184222608
// MI455X (gfx1250) — compile-verified, matrix operand order checked by reading, not run on hardware
//
#include <hip/hip_runtime.h>
#include <hip/hip_bf16.h>

typedef __attribute__((ext_vector_type(16))) __bf16 v16bf;
typedef __attribute__((ext_vector_type(8)))  __bf16 v8bf;
typedef __attribute__((ext_vector_type(8)))  float  v8f;

// ---------------------------------------------------------------------------
// WMMA helpers: NT GEMM fragments. For bf16 16x16x32 WMMA, each lane holds
// two contiguous 8-element (16B) K-chunks of its row:
//   row  = lane & 15   (A: M-row, B: N-col of B^T)
//   kb   = (lane>=16) ? 8 : 0
//   chunk0 = K [kb, kb+8),  chunk1 = K [16+kb, 16+kb+8)
// Both x/h (bf16, K-contiguous) and weights (fp32, K-contiguous, converted
// in-register) load directly into this layout — no LDS staging required.
// ---------------------------------------------------------------------------

__device__ __forceinline__ v16bf load_tile_bf16(const __bf16* rowp, int kb) {
  v8bf c0 = *(const v8bf*)(rowp + kb);
  v8bf c1 = *(const v8bf*)(rowp + kb + 16);
  return __builtin_shufflevector(c0, c1, 0,1,2,3,4,5,6,7,8,9,10,11,12,13,14,15);
}

__device__ __forceinline__ v16bf load_tile_f32(const float* rowp, int kb) {
  float4 f0 = *(const float4*)(rowp + kb);
  float4 f1 = *(const float4*)(rowp + kb + 4);
  float4 f2 = *(const float4*)(rowp + kb + 16);
  float4 f3 = *(const float4*)(rowp + kb + 20);
  v16bf a;
  a[0]=(__bf16)f0.x;  a[1]=(__bf16)f0.y;  a[2]=(__bf16)f0.z;  a[3]=(__bf16)f0.w;
  a[4]=(__bf16)f1.x;  a[5]=(__bf16)f1.y;  a[6]=(__bf16)f1.z;  a[7]=(__bf16)f1.w;
  a[8]=(__bf16)f2.x;  a[9]=(__bf16)f2.y;  a[10]=(__bf16)f2.z; a[11]=(__bf16)f2.w;
  a[12]=(__bf16)f3.x; a[13]=(__bf16)f3.y; a[14]=(__bf16)f3.z; a[15]=(__bf16)f3.w;
  return a;
}

__device__ __forceinline__ v8f wmma_bf16(v16bf a, v16bf b, v8f c) {
  return __builtin_amdgcn_wmma_f32_16x16x32_bf16(
      /*neg_a=*/false, a, /*neg_b=*/false, b,
      /*c_mod=*/(short)0, c, /*reuse_a=*/false, /*reuse_b=*/false);
}

__device__ __forceinline__ float siluf(float v) {
  return v / (1.0f + __expf(-v));
}

// ---------------------------------------------------------------------------
// Router: one wave per token. Computes 16 expert logits + shared-gate logit,
// cross-lane reduce, softmax, top-4 by 4 argmax passes, scatter dense scores.
// ---------------------------------------------------------------------------
__global__ __launch_bounds__(32)
void router_kernel(const float* __restrict__ x, const float* __restrict__ rw,
                   const float* __restrict__ wshg,
                   float* __restrict__ scores, float* __restrict__ gout,
                   int T, int D) {
  const int t = blockIdx.x;
  const int lane = threadIdx.x;
  const float* xr = x + (size_t)t * D;
  float acc[16];
  float accg = 0.0f;
#pragma unroll
  for (int e = 0; e < 16; ++e) acc[e] = 0.0f;
  for (int i = lane; i < D; i += 32) {
    float xv = xr[i];
#pragma unroll
    for (int e = 0; e < 16; ++e) acc[e] += xv * rw[e * D + i];
    accg += xv * wshg[i];
  }
#pragma unroll
  for (int s = 0; s < 5; ++s) {
    int off = 1 << s;
#pragma unroll
    for (int e = 0; e < 16; ++e) acc[e] += __shfl_xor(acc[e], off, 32);
    accg += __shfl_xor(accg, off, 32);
  }
  // softmax over 16 experts (redundant per lane, tiny)
  float mx = -3.4e38f;
#pragma unroll
  for (int e = 0; e < 16; ++e) mx = fmaxf(mx, acc[e]);
  float p[16];
  float sum = 0.0f;
#pragma unroll
  for (int e = 0; e < 16; ++e) { p[e] = __expf(acc[e] - mx); sum += p[e]; }
  float inv = 1.0f / sum;
#pragma unroll
  for (int e = 0; e < 16; ++e) p[e] *= inv;
  // top-4: find 4th-largest via 4 argmax passes on a scratch copy
  float tmp[16];
#pragma unroll
  for (int e = 0; e < 16; ++e) tmp[e] = p[e];
  float kth = 0.0f;
#pragma unroll
  for (int r = 0; r < 4; ++r) {
    float mv = -1.0f;
#pragma unroll
    for (int e = 0; e < 16; ++e) mv = fmaxf(mv, tmp[e]);
    kth = mv;
    bool cleared = false;
#pragma unroll
    for (int e = 0; e < 16; ++e) {
      if (!cleared && tmp[e] == mv) { tmp[e] = -1.0f; cleared = true; }
    }
  }
  if (lane == 0) {
#pragma unroll
    for (int e = 0; e < 16; ++e)
      scores[(size_t)t * 16 + e] = (p[e] >= kth) ? p[e] : 0.0f;
    gout[t] = 1.0f / (1.0f + __expf(-accg));
  }
}

// ---------------------------------------------------------------------------
// Deterministic per-expert compaction: ascending token order via block scan.
// ---------------------------------------------------------------------------
__global__ __launch_bounds__(256)
void compact_kernel(const float* __restrict__ scores, int* __restrict__ lists,
                    int* __restrict__ counts, int T, int E) {
  __shared__ int cnts[256];
  __shared__ int offs[256];
  const int e = blockIdx.x;
  const int tid = threadIdx.x;
  const int per = T / 256;
  const int base = tid * per;
  int c = 0;
  for (int i = 0; i < per; ++i)
    if (scores[(size_t)(base + i) * E + e] > 0.0f) ++c;
  cnts[tid] = c;
  __syncthreads();
  if (tid == 0) {
    int run = 0;
    for (int i = 0; i < 256; ++i) { offs[i] = run; run += cnts[i]; }
    counts[e] = run;
  }
  __syncthreads();
  int o = offs[tid];
  for (int i = 0; i < per; ++i) {
    int t = base + i;
    if (scores[(size_t)t * E + e] > 0.0f) lists[e * T + (o++)] = t;
  }
}

__global__ __launch_bounds__(256)
void cast_kernel(const float* __restrict__ x, __bf16* __restrict__ xb, int n) {
  int i = (blockIdx.x * blockDim.x + threadIdx.x) * 4;
  if (i + 3 < n) {
    float4 f = *(const float4*)(x + i);
    xb[i]     = (__bf16)f.x;
    xb[i + 1] = (__bf16)f.y;
    xb[i + 2] = (__bf16)f.z;
    xb[i + 3] = (__bf16)f.w;
  }
}

// ---------------------------------------------------------------------------
// Fused gate+up GEMM (NT). One wave computes a 32(M-rows)x32(F-cols) tile for
// BOTH gate and up (A loaded once, 8 WMMAs per K-step of 32). Expert mode
// (list != null): M-rows are compacted tokens of expert e, epilogue scales by
// routing score. Shared mode (list == null): dense rows, score = 1.
// Weights are fp32 in memory, converted to bf16 in-register (v_cvt).
// ---------------------------------------------------------------------------
__global__ __launch_bounds__(32)
void gateup_kernel(const __bf16* __restrict__ xb,
                   const float* __restrict__ wg_base,
                   const float* __restrict__ wu_base,
                   int D, int F, int Trows,
                   __bf16* __restrict__ hout,
                   const int* __restrict__ list_base,
                   const int* __restrict__ counts,
                   const float* __restrict__ scores, int E) {
  const int e = blockIdx.z;
  const int fbase = blockIdx.x * 32;
  const int mbase = blockIdx.y * 32;
  const int cnt = counts ? counts[e] : Trows;
  if (mbase >= cnt) return;
  const int lane = threadIdx.x & 31;
  const int r = lane & 15;
  const int half = lane >> 4;
  const int kb = half * 8;
  const int* list = list_base ? (list_base + e * Trows) : nullptr;

  int mA0 = mbase + r;
  int mA1 = mbase + 16 + r;
  int cm0 = mA0 < cnt ? mA0 : cnt - 1;
  int cm1 = mA1 < cnt ? mA1 : cnt - 1;
  int tok0 = list ? list[cm0] : cm0;
  int tok1 = list ? list[cm1] : cm1;
  const __bf16* a0p = xb + (size_t)tok0 * D;
  const __bf16* a1p = xb + (size_t)tok1 * D;
  const size_t wstride = (size_t)F * D;
  const float* bg0 = wg_base + (size_t)e * wstride + (size_t)(fbase + r) * D;
  const float* bg1 = bg0 + (size_t)16 * D;
  const float* bu0 = wu_base + (size_t)e * wstride + (size_t)(fbase + r) * D;
  const float* bu1 = bu0 + (size_t)16 * D;

  const v8f vz = {0.f, 0.f, 0.f, 0.f, 0.f, 0.f, 0.f, 0.f};
  v8f accg[2][2], accu[2][2];
#pragma unroll
  for (int i = 0; i < 2; ++i)
#pragma unroll
    for (int j = 0; j < 2; ++j) { accg[i][j] = vz; accu[i][j] = vz; }

  for (int k = 0; k < D; k += 32) {
    v16bf a0 = load_tile_bf16(a0p + k, kb);
    v16bf a1 = load_tile_bf16(a1p + k, kb);
    v16bf g0 = load_tile_f32(bg0 + k, kb);
    v16bf g1 = load_tile_f32(bg1 + k, kb);
    v16bf u0 = load_tile_f32(bu0 + k, kb);
    v16bf u1 = load_tile_f32(bu1 + k, kb);
    accg[0][0] = wmma_bf16(a0, g0, accg[0][0]);
    accg[0][1] = wmma_bf16(a0, g1, accg[0][1]);
    accg[1][0] = wmma_bf16(a1, g0, accg[1][0]);
    accg[1][1] = wmma_bf16(a1, g1, accg[1][1]);
    accu[0][0] = wmma_bf16(a0, u0, accu[0][0]);
    accu[0][1] = wmma_bf16(a0, u1, accu[0][1]);
    accu[1][0] = wmma_bf16(a1, u0, accu[1][0]);
    accu[1][1] = wmma_bf16(a1, u1, accu[1][1]);
  }

  // C layout: vgpr rr -> M = mt*16 + half*8 + rr, N = nt*16 + (lane&15)
#pragma unroll
  for (int mt = 0; mt < 2; ++mt) {
#pragma unroll
    for (int rr = 0; rr < 8; ++rr) {
      int m = mbase + mt * 16 + half * 8 + rr;
      if (m >= cnt) continue;
      int tok = list ? list[m] : m;
      float s = scores ? scores[(size_t)tok * E + e] : 1.0f;
      __bf16* hrow = hout + ((size_t)e * Trows + m) * F;
#pragma unroll
      for (int nt = 0; nt < 2; ++nt) {
        float gv = accg[mt][nt][rr];
        float uv = accu[mt][nt][rr];
        float hv = siluf(gv) * uv * s;
        hrow[fbase + nt * 16 + r] = (__bf16)hv;
      }
    }
  }
}

// ---------------------------------------------------------------------------
// Expert down-proj (NT, K = F). A = compacted h rows (bf16, already score-
// scaled), B = w_down[e] rows (fp32 -> bf16 in-register). Scatter-add each
// output element to moe_out[token] with global_atomic_add_f32.
// ---------------------------------------------------------------------------
__global__ __launch_bounds__(32)
void down_kernel(const __bf16* __restrict__ hbuf,
                 const float* __restrict__ wd_base,
                 int F, int Dout, int Trows,
                 const int* __restrict__ list_base,
                 const int* __restrict__ counts,
                 float* __restrict__ moe_out) {
  const int e = blockIdx.z;
  const int dbase = blockIdx.x * 32;
  const int mbase = blockIdx.y * 32;
  const int cnt = counts[e];
  if (mbase >= cnt) return;
  const int lane = threadIdx.x & 31;
  const int r = lane & 15;
  const int half = lane >> 4;
  const int kb = half * 8;
  const int* list = list_base + e * Trows;

  int cm0 = (mbase + r) < cnt ? (mbase + r) : cnt - 1;
  int cm1 = (mbase + 16 + r) < cnt ? (mbase + 16 + r) : cnt - 1;
  const __bf16* a0p = hbuf + ((size_t)e * Trows + cm0) * F;
  const __bf16* a1p = hbuf + ((size_t)e * Trows + cm1) * F;
  const float* b0 = wd_base + (size_t)e * Dout * F + (size_t)(dbase + r) * F;
  const float* b1 = b0 + (size_t)16 * F;

  const v8f vz = {0.f, 0.f, 0.f, 0.f, 0.f, 0.f, 0.f, 0.f};
  v8f acc[2][2];
#pragma unroll
  for (int i = 0; i < 2; ++i)
#pragma unroll
    for (int j = 0; j < 2; ++j) acc[i][j] = vz;

  for (int k = 0; k < F; k += 32) {
    v16bf a0 = load_tile_bf16(a0p + k, kb);
    v16bf a1 = load_tile_bf16(a1p + k, kb);
    v16bf w0 = load_tile_f32(b0 + k, kb);
    v16bf w1 = load_tile_f32(b1 + k, kb);
    acc[0][0] = wmma_bf16(a0, w0, acc[0][0]);
    acc[0][1] = wmma_bf16(a0, w1, acc[0][1]);
    acc[1][0] = wmma_bf16(a1, w0, acc[1][0]);
    acc[1][1] = wmma_bf16(a1, w1, acc[1][1]);
  }

#pragma unroll
  for (int mt = 0; mt < 2; ++mt) {
#pragma unroll
    for (int rr = 0; rr < 8; ++rr) {
      int m = mbase + mt * 16 + half * 8 + rr;
      if (m >= cnt) continue;
      int tok = list[m];
      float* orow = moe_out + (size_t)tok * Dout;
#pragma unroll
      for (int nt = 0; nt < 2; ++nt)
        atomicAdd(&orow[dbase + nt * 16 + r], acc[mt][nt][rr]);
    }
  }
}

// ---------------------------------------------------------------------------
// Shared-expert down-proj + final combine: out = moe_out + g[t] * (h @ Wd^T).
// ---------------------------------------------------------------------------
__global__ __launch_bounds__(32)
void down_shared_kernel(const __bf16* __restrict__ hsh,
                        const float* __restrict__ wsd,
                        int FS, int Dout,
                        const float* __restrict__ moe_out,
                        const float* __restrict__ g,
                        float* __restrict__ out) {
  const int dbase = blockIdx.x * 32;
  const int tbase = blockIdx.y * 32;
  const int lane = threadIdx.x & 31;
  const int r = lane & 15;
  const int half = lane >> 4;
  const int kb = half * 8;

  const __bf16* a0p = hsh + (size_t)(tbase + r) * FS;
  const __bf16* a1p = a0p + (size_t)16 * FS;
  const float* b0 = wsd + (size_t)(dbase + r) * FS;
  const float* b1 = b0 + (size_t)16 * FS;

  const v8f vz = {0.f, 0.f, 0.f, 0.f, 0.f, 0.f, 0.f, 0.f};
  v8f acc[2][2];
#pragma unroll
  for (int i = 0; i < 2; ++i)
#pragma unroll
    for (int j = 0; j < 2; ++j) acc[i][j] = vz;

  for (int k = 0; k < FS; k += 32) {
    v16bf a0 = load_tile_bf16(a0p + k, kb);
    v16bf a1 = load_tile_bf16(a1p + k, kb);
    v16bf w0 = load_tile_f32(b0 + k, kb);
    v16bf w1 = load_tile_f32(b1 + k, kb);
    acc[0][0] = wmma_bf16(a0, w0, acc[0][0]);
    acc[0][1] = wmma_bf16(a0, w1, acc[0][1]);
    acc[1][0] = wmma_bf16(a1, w0, acc[1][0]);
    acc[1][1] = wmma_bf16(a1, w1, acc[1][1]);
  }

#pragma unroll
  for (int mt = 0; mt < 2; ++mt) {
#pragma unroll
    for (int rr = 0; rr < 8; ++rr) {
      int t = tbase + mt * 16 + half * 8 + rr;
      float gv = g[t];
#pragma unroll
      for (int nt = 0; nt < 2; ++nt) {
        int col = dbase + nt * 16 + r;
        size_t idx = (size_t)t * Dout + col;
        out[idx] = moe_out[idx] + gv * acc[mt][nt][rr];
      }
    }
  }
}

// ---------------------------------------------------------------------------
extern "C" void kernel_launch(void* const* d_in, const int* in_sizes, int n_in,
                              void* d_out, int out_size, void* d_ws, size_t ws_size,
                              hipStream_t stream) {
  (void)in_sizes; (void)n_in; (void)out_size; (void)ws_size;
  const float* x    = (const float*)d_in[0];  // [T, D]
  const float* rw   = (const float*)d_in[1];  // [E, D]
  const float* wg   = (const float*)d_in[2];  // [E, F, D]
  const float* wu   = (const float*)d_in[3];  // [E, F, D]
  const float* wd   = (const float*)d_in[4];  // [E, D, F]
  const float* wsg  = (const float*)d_in[5];  // [FS, D]
  const float* wsu  = (const float*)d_in[6];  // [FS, D]
  const float* wsd  = (const float*)d_in[7];  // [D, FS]
  const float* wshg = (const float*)d_in[8];  // [1, D]
  float* out = (float*)d_out;

  const int T = 2048, D = 2048, E = 16, F = 1408, FS = 5632;

  char* ws = (char*)d_ws;
  size_t off = 0;
  auto alloc = [&](size_t bytes) -> char* {
    char* p = ws + off;
    off = (off + bytes + 255) & ~(size_t)255;
    return p;
  };
  float*  scores = (float*)alloc((size_t)T * E * 4);
  float*  gbuf   = (float*)alloc((size_t)T * 4);
  int*    counts = (int*)alloc((size_t)E * 4);
  int*    lists  = (int*)alloc((size_t)E * T * 4);
  __bf16* xb     = (__bf16*)alloc((size_t)T * D * 2);
  float*  moe    = (float*)alloc((size_t)T * D * 4);
  __bf16* hcomp  = (__bf16*)alloc((size_t)E * T * F * 2);
  __bf16* hsh    = (__bf16*)alloc((size_t)T * FS * 2);

  hipMemsetAsync(moe, 0, (size_t)T * D * sizeof(float), stream);

  router_kernel<<<T, 32, 0, stream>>>(x, rw, wshg, scores, gbuf, T, D);
  compact_kernel<<<E, 256, 0, stream>>>(scores, lists, counts, T, E);
  cast_kernel<<<(T * D) / (256 * 4), 256, 0, stream>>>(x, xb, T * D);

  dim3 gup(F / 32, T / 32, E);
  gateup_kernel<<<gup, 32, 0, stream>>>(xb, wg, wu, D, F, T, hcomp,
                                        lists, counts, scores, E);
  dim3 gsh(FS / 32, T / 32, 1);
  gateup_kernel<<<gsh, 32, 0, stream>>>(xb, wsg, wsu, D, FS, T, hsh,
                                        nullptr, nullptr, nullptr, E);
  dim3 gdn(D / 32, T / 32, E);
  down_kernel<<<gdn, 32, 0, stream>>>(hcomp, wd, F, D, T, lists, counts, moe);
  dim3 gds(D / 32, T / 32, 1);
  down_shared_kernel<<<gds, 32, 0, stream>>>(hsh, wsd, FS, D, moe, gbuf, out);
}
